// MMSingleStreamBlock_11957188952653
// MI455X (gfx1250) — matrix operand order repaired and checked by reading, not compiled, not run on hardware
//
#include <hip/hip_runtime.h>

// ---------------- problem constants ----------------
#define LQ   2304
#define HID  3072
#define NH   24
#define HD   128
#define IMGL 2048
#define MLPD 12288
#define N1   21504          // 3*HID + MLP
#define N2   15360          // HID + MLP
#define QKV  9216           // 3*HID
#define EPSF 1e-6f

// ---------------- CDNA5 async-to-LDS availability --------------------------
#if defined(__AMDGCN__) && __has_builtin(__builtin_amdgcn_global_load_async_to_lds_b128)
#define ASYNC_OK 1
#else
#define ASYNC_OK 0
#endif

#define AS1 __attribute__((address_space(1)))
#define AS3 __attribute__((address_space(3)))

typedef __attribute__((ext_vector_type(4))) int v4i;

template <int N>
__device__ inline void wait_async() {
#if ASYNC_OK
#if __has_builtin(__builtin_amdgcn_s_wait_asynccnt)
    __builtin_amdgcn_s_wait_asynccnt(N);
#else
    asm volatile("s_wait_asynccnt %0" :: "i"(N) : "memory");
#endif
#endif
}

// ---------------- types ----------------
typedef __attribute__((ext_vector_type(16))) __bf16         bf16x16;
typedef __attribute__((ext_vector_type(8)))  float          f32x8;
typedef __attribute__((ext_vector_type(8)))  unsigned short us8;

union Frag16 { bf16x16 bf; us8 us[2]; };

__device__ inline unsigned short f2bfu(float f) {
    unsigned u = __float_as_uint(f);
    unsigned r = u + 0x7FFFu + ((u >> 16) & 1u);   // round-to-nearest-even
    return (unsigned short)(r >> 16);
}

__device__ inline f32x8 zero8() {
    f32x8 z;
#pragma unroll
    for (int i = 0; i < 8; ++i) z[i] = 0.f;
    return z;
}

__device__ inline f32x8 wmma_bf16(bf16x16 a, bf16x16 b, f32x8 c) {
    return __builtin_amdgcn_wmma_f32_16x16x32_bf16(false, a, false, b,
                                                   (short)0, c, false, false);
}

// A-matrix fragment (16x32, row-major src, stride ld in elements).
// Lane l<16: row=l, K = 0..7 and 16..23 ; lane>=16: K = 8..15 and 24..31.
__device__ inline bf16x16 load_fragA(const unsigned short* base, int ld) {
    int lane = threadIdx.x & 31;
    const unsigned short* p = base + (size_t)(lane & 15) * ld + ((lane >> 4) << 3);
    Frag16 f;
    f.us[0] = *(const us8*)p;
    f.us[1] = *(const us8*)(p + 16);
    return f.bf;
}

// B-matrix fragment, source stored transposed as [N][K] row-major (ld = K).
// Lane l<16: col n=l, K = 0..7 and 16..23 ; lane>=16: K = 8..15 and 24..31.
__device__ inline bf16x16 load_fragB(const unsigned short* base, int ld) {
    int lane = threadIdx.x & 31;
    const unsigned short* p = base + (size_t)(lane & 15) * ld + ((lane >> 4) << 3);
    Frag16 f;
    f.us[0] = *(const us8*)p;
    f.us[1] = *(const us8*)(p + 16);
    return f.bf;
}

// ---------------- kernel 0: fp32 -> bf16 transpose ([R][C] -> [C][R]) ------
__global__ __launch_bounds__(256) void k_transpose_f32_bf16(
    const float* __restrict__ in, unsigned short* __restrict__ out, int R, int C) {
    __shared__ float tile[32][33];
    int c0 = blockIdx.x * 32, r0 = blockIdx.y * 32;
    int tx = threadIdx.x & 31, ty = threadIdx.x >> 5;     // ty: 0..7
#pragma unroll
    for (int i = 0; i < 4; ++i)
        tile[ty + i * 8][tx] = in[(size_t)(r0 + ty + i * 8) * C + c0 + tx];
    __syncthreads();
#pragma unroll
    for (int i = 0; i < 4; ++i)
        out[(size_t)(c0 + ty + i * 8) * R + r0 + tx] = f2bfu(tile[tx][ty + i * 8]);
}

// ---------------- kernel 1: mod = silu(vec) @ mod_w + mod_b ---------------
__global__ __launch_bounds__(256) void k_mod_gemv(
    const float* __restrict__ vec, const float* __restrict__ W,
    const float* __restrict__ b, float* __restrict__ mod) {
    __shared__ float sv[HID];
    for (int i = threadIdx.x; i < HID; i += 256) {
        float v = vec[i];
        sv[i] = v / (1.f + __expf(-v));
    }
    __syncthreads();
    int j = blockIdx.x * 256 + threadIdx.x;     // < QKV
    float acc = 0.f;
    for (int i = 0; i < HID; ++i) acc = fmaf(sv[i], W[(size_t)i * QKV + j], acc);
    mod[j] = acc + b[j];
}

// ---------------- kernel 2: layernorm + modulate -> bf16 ------------------
__global__ __launch_bounds__(256) void k_ln_mod(
    const float* __restrict__ x, const float* __restrict__ mod,
    unsigned short* __restrict__ xmodbf) {
    int t = blockIdx.x;
    const float* row = x + (size_t)t * HID;
    float s = 0.f, s2 = 0.f;
    for (int c = threadIdx.x; c < HID; c += 256) {
        float v = row[c];
        s += v; s2 += v * v;
    }
#pragma unroll
    for (int m = 1; m < 32; m <<= 1) {
        s  += __shfl_xor(s,  m, 32);
        s2 += __shfl_xor(s2, m, 32);
    }
    __shared__ float sh[2][8];
    int wv = threadIdx.x >> 5;
    if ((threadIdx.x & 31) == 0) { sh[0][wv] = s; sh[1][wv] = s2; }
    __syncthreads();
    float ts = 0.f, ts2 = 0.f;
#pragma unroll
    for (int i = 0; i < 8; ++i) { ts += sh[0][i]; ts2 += sh[1][i]; }
    float mu  = ts / (float)HID;
    float var = ts2 / (float)HID - mu * mu;
    float rs  = rsqrtf(var + EPSF);
    for (int c = threadIdx.x; c < HID; c += 256) {
        float v = (row[c] - mu) * rs;
        v = v * (1.f + mod[HID + c]) + mod[c];      // shift=mod[0:], scale=mod[HID:]
        xmodbf[(size_t)t * HID + c] = f2bfu(v);
    }
}

// ---------------- kernel 3/7: WMMA GEMM, A[M][K]bf16 x Bt[N][K]bf16 -------
// Double-buffered LDS A-tile, async-to-LDS staging (ASYNCcnt) when available.
// MODE 0: C = A*B + bias               (fp32 out)
// MODE 1: C = resid + (A*B+bias)*gate  (final output)
template <int MODE>
__global__ __launch_bounds__(256) void k_gemm_bf16(
    const unsigned short* __restrict__ A, const unsigned short* __restrict__ Bt,
    const float* __restrict__ bias, float* __restrict__ C,
    const float* __restrict__ resid, const float* __restrict__ gate,
    int M, int N, int K) {
    __shared__ __attribute__((aligned(16))) unsigned short sA[2][128][40]; // 32+8 pad
    int tid  = threadIdx.x;
    int wave = tid >> 5, lane = tid & 31;
    int bm = blockIdx.y * 128, bn = blockIdx.x * 128;
    int wm = (wave >> 2) * 64, wn = (wave & 3) * 32;

    // staging: 128x32 bf16 tile, 2 x 16B per thread (2 async instrs / wave)
    auto stage = [&](int k0, int buf) {
#pragma unroll
        for (int it = 0; it < 2; ++it) {
            int idx = tid + it * 256;             // 0..511
            int row = idx >> 2, chunk = idx & 3;  // 4 chunks of 8 bf16
            const unsigned short* src = A + (size_t)(bm + row) * K + k0 + chunk * 8;
#if ASYNC_OK
            __builtin_amdgcn_global_load_async_to_lds_b128(
                (AS1 v4i*)src,
                (AS3 v4i*)&sA[buf][row][chunk * 8], 0, 0);
#else
            *(us8*)(&sA[buf][row][chunk * 8]) = *(const us8*)src;
#endif
        }
    };

    f32x8 acc[4][2];
#pragma unroll
    for (int i = 0; i < 4; ++i)
#pragma unroll
        for (int j = 0; j < 2; ++j) acc[i][j] = zero8();

    int nk = K >> 5;
    stage(0, 0);
    for (int t = 0; t < nk; ++t) {
        int buf = t & 1;
        int k0  = t << 5;
        if (t + 1 < nk) {
            stage(k0 + 32, buf ^ 1);   // async-fetch next tile into other buffer
            wait_async<2>();           // drain only the older tile (in-order)
        } else {
            wait_async<0>();
        }
        __syncthreads();               // tile t visible to all waves

        // prefetch next B rows toward the WGP while computing
        if (t + 1 < nk)
            __builtin_prefetch((const void*)(Bt + (size_t)(bn + wn + (lane & 15)) * K + k0 + 32), 0, 3);

        bf16x16 bfrag[2];
#pragma unroll
        for (int j = 0; j < 2; ++j)
            bfrag[j] = load_fragB(Bt + (size_t)(bn + wn + j * 16) * K + k0, K);

#pragma unroll
        for (int i = 0; i < 4; ++i) {
            bf16x16 afrag = load_fragA(&sA[buf][wm + i * 16][0], 40);
#pragma unroll
            for (int j = 0; j < 2; ++j)
                acc[i][j] = wmma_bf16(afrag, bfrag[j], acc[i][j]);
        }
        __syncthreads();               // all waves done reading buf before overwrite
    }

    // epilogue: C layout — VGPR r, lane l: row = r + 8*(l>>4), col = l&15
    int coln = lane & 15, half = lane >> 4;
#pragma unroll
    for (int i = 0; i < 4; ++i) {
#pragma unroll
        for (int j = 0; j < 2; ++j) {
            int n = bn + wn + j * 16 + coln;
            float bv = bias[n];
            float gv = (MODE == 1) ? gate[n] : 0.f;
#pragma unroll
            for (int r = 0; r < 8; ++r) {
                int m = bm + wm + i * 16 + half * 8 + r;
                size_t off = (size_t)m * N + n;
                float v = acc[i][j][r] + bv;
                if (MODE == 1) v = resid[off] + v * gv;
                C[off] = v;
            }
        }
    }
}

// ---------------- kernel 4: rmsnorm(q,k) + rope -> bf16 head-major --------
__global__ __launch_bounds__(128) void k_qkv_post(
    const float* __restrict__ proj,
    const float* __restrict__ qw, const float* __restrict__ kw,
    const float* __restrict__ fc, const float* __restrict__ fs,
    unsigned short* __restrict__ qb, unsigned short* __restrict__ kb,
    unsigned short* __restrict__ vbT) {
    int t = blockIdx.x, h = blockIdx.y, d = threadIdx.x;
    const float* p = proj + (size_t)t * N1 + h * HD;
    float q = p[d], k = p[HID + d], v = p[2 * HID + d];

    float sq = q * q, sk = k * k;
#pragma unroll
    for (int m = 1; m < 32; m <<= 1) {
        sq += __shfl_xor(sq, m, 32);
        sk += __shfl_xor(sk, m, 32);
    }
    __shared__ float shq[4], shk[4];
    int wv = threadIdx.x >> 5;
    if ((threadIdx.x & 31) == 0) { shq[wv] = sq; shk[wv] = sk; }
    __syncthreads();
    sq = shq[0] + shq[1] + shq[2] + shq[3];
    sk = shk[0] + shk[1] + shk[2] + shk[3];
    float qn = q * rsqrtf(sq / (float)HD + EPSF) * qw[d];
    float kn = k * rsqrtf(sk / (float)HD + EPSF) * kw[d];

    if (t < IMGL) {   // interleaved RoPE on image tokens
        float q2 = __shfl_xor(qn, 1, 32), k2 = __shfl_xor(kn, 1, 32);
        float c = fc[(size_t)t * HD + (d & ~1)];
        float s = fs[(size_t)t * HD + (d & ~1)];
        if ((d & 1) == 0) { qn = qn * c - q2 * s; kn = kn * c - k2 * s; }
        else              { qn = q2 * s + qn * c; kn = k2 * s + kn * c; }
    }
    size_t base = ((size_t)h * LQ + t) * HD + d;
    qb[base] = f2bfu(qn);
    kb[base] = f2bfu(kn);
    vbT[((size_t)h * HD + d) * LQ + t] = f2bfu(v);   // V transposed [NH][HD][L]
}

// ---------------- kernel 5: gelu(mlp) -> a2 cols [HID, N2) ----------------
__global__ __launch_bounds__(256) void k_gelu(
    const float* __restrict__ proj, unsigned short* __restrict__ a2) {
    size_t idx = (size_t)blockIdx.x * 256 + threadIdx.x;
    if (idx >= (size_t)LQ * MLPD) return;
    int t = (int)(idx / MLPD), j = (int)(idx % MLPD);
    float v = proj[(size_t)t * N1 + QKV + j];
    float g = 0.5f * v * (1.f + tanhf(0.7978845608f * (v + 0.044715f * v * v * v)));
    a2[(size_t)t * N2 + HID + j] = f2bfu(g);
}

// ---------------- kernel 6: flash attention (bf16 WMMA) -------------------
// grid (L/64, NH), 128 threads = 4 waves, 16 query rows per wave
__global__ __launch_bounds__(128) void k_attn(
    const unsigned short* __restrict__ Q,   // [NH][L][HD]
    const unsigned short* __restrict__ Kt,  // [NH][L][HD]
    const unsigned short* __restrict__ Vt,  // [NH][HD][L]
    unsigned short* __restrict__ Out,       // a2: [L][N2], cols [0,HID)
    float scale) {
    int head = blockIdx.y;
    int wave = threadIdx.x >> 5, lane = threadIdx.x & 31;
    int coln = lane & 15, half = lane >> 4;
    int q0 = blockIdx.x * 64 + wave * 16;

    const unsigned short* Qh = Q  + ((size_t)head * LQ + q0) * HD;
    const unsigned short* Kh = Kt + (size_t)head * LQ * HD;
    const unsigned short* Vh = Vt + (size_t)head * HD * LQ;

    bf16x16 qf[4];
#pragma unroll
    for (int kk = 0; kk < 4; ++kk) qf[kk] = load_fragA(Qh + kk * 32, HD);

    f32x8 o[8];
#pragma unroll
    for (int t8 = 0; t8 < 8; ++t8) o[t8] = zero8();
    float mrow[8], lrow[8];
#pragma unroll
    for (int r = 0; r < 8; ++r) { mrow[r] = -3.0e38f; lrow[r] = 0.f; }

    __shared__ __attribute__((aligned(16))) unsigned short sP[4][16][40];

    for (int k0 = 0; k0 < LQ; k0 += 32) {
        f32x8 s0 = zero8(), s1 = zero8();
#pragma unroll
        for (int kk = 0; kk < 4; ++kk) {
            bf16x16 kf0 = load_fragB(Kh + (size_t)k0 * HD + kk * 32, HD);
            bf16x16 kf1 = load_fragB(Kh + (size_t)(k0 + 16) * HD + kk * 32, HD);
            s0 = wmma_bf16(qf[kk], kf0, s0);
            s1 = wmma_bf16(qf[kk], kf1, s1);
        }
        __syncthreads();
        // online softmax: row = r + 8*half, reduce across the 16-lane half
#pragma unroll
        for (int r = 0; r < 8; ++r) {
            float a = s0[r] * scale, b = s1[r] * scale;
            float mx = fmaxf(a, b);
#pragma unroll
            for (int m = 1; m < 16; m <<= 1) mx = fmaxf(mx, __shfl_xor(mx, m, 32));
            float mnew = fmaxf(mrow[r], mx);
            float corr = __expf(mrow[r] - mnew);
            mrow[r] = mnew;
            float p0 = __expf(a - mnew), p1 = __expf(b - mnew);
            float ps = p0 + p1;
#pragma unroll
            for (int m = 1; m < 16; m <<= 1) ps += __shfl_xor(ps, m, 32);
            lrow[r] = lrow[r] * corr + ps;
#pragma unroll
            for (int t8 = 0; t8 < 8; ++t8) o[t8][r] *= corr;
            sP[wave][half * 8 + r][coln]      = f2bfu(p0);
            sP[wave][half * 8 + r][16 + coln] = f2bfu(p1);
        }
        __syncthreads();
        bf16x16 pf = load_fragA(&sP[wave][0][0], 40);
#pragma unroll
        for (int t8 = 0; t8 < 8; ++t8) {
            bf16x16 vf = load_fragB(Vh + (size_t)(t8 * 16) * LQ + k0, LQ);
            o[t8] = wmma_bf16(pf, vf, o[t8]);
        }
    }

#pragma unroll
    for (int t8 = 0; t8 < 8; ++t8) {
#pragma unroll
        for (int r = 0; r < 8; ++r) {
            int m = q0 + half * 8 + r;
            int d = t8 * 16 + coln;
            Out[(size_t)m * N2 + head * HD + d] = f2bfu(o[t8][r] / lrow[r]);
        }
    }
}

// ---------------- launch ---------------------------------------------------
extern "C" void kernel_launch(void* const* d_in, const int* in_sizes, int n_in,
                              void* d_out, int out_size, void* d_ws, size_t ws_size,
                              hipStream_t stream) {
    (void)in_sizes; (void)n_in; (void)out_size; (void)ws_size;

    const float* x     = (const float*)d_in[0];
    const float* vec   = (const float*)d_in[1];
    /* d_in[2] img_len == IMGL (compile-time) */
    const float* fc    = (const float*)d_in[3];
    const float* fs    = (const float*)d_in[4];
    const float* mod_w = (const float*)d_in[5];
    const float* mod_b = (const float*)d_in[6];
    const float* w1    = (const float*)d_in[7];
    const float* b1    = (const float*)d_in[8];
    const float* qw    = (const float*)d_in[9];
    const float* kw    = (const float*)d_in[10];
    const float* w2    = (const float*)d_in[11];
    const float* b2    = (const float*)d_in[12];
    float* out = (float*)d_out;

    char* p = (char*)d_ws;
    float*          mod    = (float*)p;          p += (size_t)QKV * 4;
    unsigned short* xmodbf = (unsigned short*)p; p += (size_t)LQ * HID * 2;
    unsigned short* w1t    = (unsigned short*)p; p += (size_t)N1 * HID * 2;
    float*          proj   = (float*)p;          p += (size_t)LQ * N1 * 4;
    unsigned short* qb     = (unsigned short*)p; p += (size_t)NH * LQ * HD * 2;
    unsigned short* kb     = (unsigned short*)p; p += (size_t)NH * LQ * HD * 2;
    unsigned short* vbT    = (unsigned short*)p; p += (size_t)NH * LQ * HD * 2;
    unsigned short* w2t    = (unsigned short*)p; p += (size_t)HID * N2 * 2;
    unsigned short* a2     = w1t;                // alias: w1t dead after GEMM1

    // weight conversion/transpose to bf16 [N][K]
    k_transpose_f32_bf16<<<dim3(N1 / 32, HID / 32), 256, 0, stream>>>(w1, w1t, HID, N1);
    k_transpose_f32_bf16<<<dim3(HID / 32, N2 / 32), 256, 0, stream>>>(w2, w2t, N2, HID);

    // modulation vector
    k_mod_gemv<<<QKV / 256, 256, 0, stream>>>(vec, mod_w, mod_b, mod);

    // layernorm + modulate -> bf16
    k_ln_mod<<<LQ, 256, 0, stream>>>(x, mod, xmodbf);

    // GEMM1: proj = x_mod @ w1 + b1
    k_gemm_bf16<0><<<dim3(N1 / 128, LQ / 128), 256, 0, stream>>>(
        xmodbf, w1t, b1, proj, nullptr, nullptr, LQ, N1, HID);

    // q/k rmsnorm + rope, v transpose
    k_qkv_post<<<dim3(LQ, NH), 128, 0, stream>>>(proj, qw, kw, fc, fs, qb, kb, vbT);

    // gelu(mlp) into a2
    size_t ng = (size_t)LQ * MLPD;
    k_gelu<<<(unsigned)((ng + 255) / 256), 256, 0, stream>>>(proj, a2);

    // attention into a2 cols [0, HID)
    k_attn<<<dim3(LQ / 64, NH), 128, 0, stream>>>(qb, kb, vbT, a2, 0.08838834764831845f);

    // GEMM2 + bias + gate + residual -> out
    k_gemm_bf16<1><<<dim3(HID / 128, LQ / 128), 256, 0, stream>>>(
        a2, w2t, b2, out, x, mod + 2 * HID, LQ, HID, N2);
}
